// LlamaAttention_42460046688396
// MI455X (gfx1250) — matrix operand order repaired and checked by reading, not compiled, not run on hardware
//
#include <hip/hip_runtime.h>

// ---------- types ----------
typedef __bf16 bf16;
typedef __attribute__((ext_vector_type(16))) __bf16 v16bf;
typedef __attribute__((ext_vector_type(8)))  __bf16 bf16x8;
typedef __attribute__((ext_vector_type(4)))  __bf16 bf16x4;
typedef __attribute__((ext_vector_type(8)))  float  v8f;
typedef __attribute__((ext_vector_type(4)))  float  f32x4;

union V16 { v16bf v; bf16x8 h[2]; };

static __device__ __forceinline__ v8f wmma_bf16(const V16& a, const V16& b, v8f c) {
  return __builtin_amdgcn_wmma_f32_16x16x32_bf16(false, a.v, false, b.v, (short)0, c, false, false);
}
static __device__ __forceinline__ v8f zero8() {
  v8f z = {0.f,0.f,0.f,0.f,0.f,0.f,0.f,0.f};
  return z;
}
static __device__ __forceinline__ float rmax16(float v) {
#pragma unroll
  for (int off = 1; off < 16; off <<= 1) v = fmaxf(v, __shfl_xor(v, off, 32));
  return v;
}
static __device__ __forceinline__ float rsum16(float v) {
#pragma unroll
  for (int off = 1; off < 16; off <<= 1) v += __shfl_xor(v, off, 32);
  return v;
}
// async global->LDS copy of 16 bytes (8 bf16). lds_off = low 32 bits of generic LDS ptr.
static __device__ __forceinline__ void async_cp16(unsigned lds_off, const bf16* g) {
  asm volatile("global_load_async_to_lds_b128 %0, %1, off" :: "v"(lds_off), "v"(g) : "memory");
}

// ---------- constants ----------
#define BQ 2
#define SQ 2048
#define HID 4096
#define NH 32
#define NKV 8
#define DH 128

// ================== bf16 WMMA GEMM, async-DMA double-buffered, compile-time shapes ==================
// C[M,N] = A[M,K] @ BT[N,K]^T.  A, BT bf16 row-major; C fp32.
// Block tile 128(M) x 128(N), K-step 64. 256 threads = 8 waves (4M x 2N); wave tile 32x64.
#define GM 128
#define GN 128
#define GK 64

template <int N, int K>
__global__ __launch_bounds__(256)
void gemm_bf16_async(const bf16* __restrict__ A, const bf16* __restrict__ BT,
                     float* __restrict__ C) {
  __shared__ bf16 sA[2][GM][GK];   // 2 x 16 KB
  __shared__ bf16 sB[2][GN][GK];   // 2 x 16 KB

  const int tid  = threadIdx.x;
  const int wave = tid >> 5;
  const int lane = tid & 31;
  const int ln   = lane & 15;
  const int hi   = lane >> 4;
  const int wm   = (wave & 3) * 32;   // wave M offset in tile
  const int wn   = (wave >> 2) * 64;  // wave N offset in tile
  const int mblk = blockIdx.y * GM;
  const int nblk = blockIdx.x * GN;

  v8f acc[2][4];
#pragma unroll
  for (int i = 0; i < 2; i++)
#pragma unroll
    for (int t = 0; t < 4; t++) acc[i][t] = zero8();

  // issue one K-block stage: 8 async b128 per thread (4 for A, 4 for B)
  auto issue = [&](int buf, int kb) {
#pragma unroll
    for (int r = 0; r < 4; r++) {
      int chunk = tid + r * 256;          // 0..1023
      int m  = chunk >> 3;                // row in tile
      int c8 = (chunk & 7) << 3;          // bf16 element offset (16B chunks)
      async_cp16((unsigned)(size_t)&sA[buf][m][c8],
                 A + (size_t)(mblk + m) * K + kb + c8);
    }
#pragma unroll
    for (int r = 0; r < 4; r++) {
      int chunk = tid + r * 256;
      int n  = chunk >> 3;
      int c8 = (chunk & 7) << 3;
      async_cp16((unsigned)(size_t)&sB[buf][n][c8],
                 BT + (size_t)(nblk + n) * K + kb + c8);
    }
  };

  const int nkb = K / GK;
  issue(0, 0);
  for (int i = 0; i < nkb; i++) {
    const int buf = i & 1;
    if (i + 1 < nkb) {
      issue(buf ^ 1, (i + 1) * GK);
      asm volatile("s_wait_asynccnt 0x8" ::: "memory");  // drain current tile only
    } else {
      asm volatile("s_wait_asynccnt 0x0" ::: "memory");
    }
    __syncthreads();

#pragma unroll
    for (int ks = 0; ks < 2; ks++) {      // two 32-deep WMMA k-steps
      const int k0 = ks * 32;
      V16 a0, a1;
      a0.h[0] = *(const bf16x8*)&sA[buf][wm + ln][k0 + 8 * hi];
      a0.h[1] = *(const bf16x8*)&sA[buf][wm + ln][k0 + 16 + 8 * hi];
      a1.h[0] = *(const bf16x8*)&sA[buf][wm + 16 + ln][k0 + 8 * hi];
      a1.h[1] = *(const bf16x8*)&sA[buf][wm + 16 + ln][k0 + 16 + 8 * hi];
#pragma unroll
      for (int nt = 0; nt < 4; nt++) {
        V16 bfg;
        bfg.h[0] = *(const bf16x8*)&sB[buf][wn + nt * 16 + ln][k0 + 8 * hi];
        bfg.h[1] = *(const bf16x8*)&sB[buf][wn + nt * 16 + ln][k0 + 16 + 8 * hi];
        acc[0][nt] = wmma_bf16(a0, bfg, acc[0][nt]);
        acc[1][nt] = wmma_bf16(a1, bfg, acc[1][nt]);
      }
    }
    __syncthreads();
  }

  // epilogue (C layout: M = r + 8*hi, N = ln); N is compile-time -> immediate strides
#pragma unroll
  for (int mi = 0; mi < 2; mi++)
#pragma unroll
    for (int nt = 0; nt < 4; nt++) {
      float* cp = C + (size_t)(mblk + wm + mi * 16 + 8 * hi) * N + nblk + wn + nt * 16 + ln;
#pragma unroll
      for (int r = 0; r < 8; r++) cp[(size_t)r * N] = acc[mi][nt][r];
    }
}

// ================== fp32 -> bf16 convert (vectorized) ==================
__global__ __launch_bounds__(256)
void convert_bf16(const float* __restrict__ X, bf16* __restrict__ Y, int n4) {
  int i = blockIdx.x * blockDim.x + threadIdx.x;
  if (i >= n4) return;
  f32x4 v = ((const f32x4*)X)[i];
  bf16x4 w;
#pragma unroll
  for (int j = 0; j < 4; j++) w[j] = (bf16)v[j];
  ((bf16x4*)Y)[i] = w;
}

// ================== fp32 W[R,Ncol] -> bf16 WT[Ncol,R] tiled transpose ==================
__global__ __launch_bounds__(256)
void transpose_convert(const float* __restrict__ W, bf16* __restrict__ WT, int R, int Ncol) {
  __shared__ float t[32][33];
  int tx = threadIdx.x & 31, ty = threadIdx.x >> 5;
  int c0 = blockIdx.x * 32, r0 = blockIdx.y * 32;
#pragma unroll
  for (int j = 0; j < 32; j += 8)
    t[ty + j][tx] = W[(size_t)(r0 + ty + j) * Ncol + c0 + tx];
  __syncthreads();
#pragma unroll
  for (int j = 0; j < 32; j += 8)
    WT[(size_t)(c0 + ty + j) * R + r0 + tx] = (bf16)t[tx][ty + j];
}

// ================== bf16 [BH][S][D] -> [BH][D][S] transpose (for V) ==================
__global__ __launch_bounds__(256)
void transpose_v(const bf16* __restrict__ X, bf16* __restrict__ Y) {
  __shared__ bf16 t[32][33];
  int bh = blockIdx.z;
  int d0 = blockIdx.x * 32, s0 = blockIdx.y * 32;
  const bf16* Xp = X + (size_t)bh * SQ * DH;
  bf16*       Yp = Y + (size_t)bh * DH * SQ;
  int tx = threadIdx.x & 31, ty = threadIdx.x >> 5;
#pragma unroll
  for (int j = 0; j < 32; j += 8)
    t[ty + j][tx] = Xp[(size_t)(s0 + ty + j) * DH + d0 + tx];
  __syncthreads();
#pragma unroll
  for (int j = 0; j < 32; j += 8)
    Yp[(size_t)(d0 + ty + j) * SQ + s0 + tx] = t[tx][ty + j];
}

// ================== RoPE + head split (fp32 qkv -> bf16 Q/K/V) ==================
// NOTE: 1/sqrt(D) attention scale is folded into Q here.
__global__ __launch_bounds__(256)
void rope_split(const int* __restrict__ positions, const float* __restrict__ qkv,
                bf16* __restrict__ Qb, bf16* __restrict__ Kb, bf16* __restrict__ Vb,
                int total) {
  int idx = blockIdx.x * blockDim.x + threadIdx.x;
  if (idx >= total) return;
  const int CW = (NH + 2 * NKV) * DH;  // 6144
  int bs = idx / CW;
  int col = idx - bs * CW;
  int b = bs / SQ;
  int s = bs - b * SQ;
  float x = qkv[(size_t)bs * CW + col];

  if (col < NH * DH) {                       // Q, rotate + scale
    int h = col >> 7, d = col & 127, i = d & 63;
    float pos  = (float)positions[bs];
    float freq = __expf(-(float)i * 0.14391156731f);   // ln(1e4)/64
    float ang  = pos * freq;
    float c = __cosf(ang), sn = __sinf(ang);
    float xp = qkv[(size_t)bs * CW + (col ^ 64)];
    float o = (d < 64) ? (x * c - xp * sn) : (x * c + xp * sn);
    o *= 0.08838834764831845f;  // 1/sqrt(128) folded into Q
    Qb[((size_t)(b * NH + h) * SQ + s) * DH + d] = (bf16)o;
  } else if (col < (NH + NKV) * DH) {        // K, rotate
    int c2 = col - NH * DH;
    int kv = c2 >> 7, d = c2 & 127, i = d & 63;
    float pos  = (float)positions[bs];
    float freq = __expf(-(float)i * 0.14391156731f);
    float ang  = pos * freq;
    float c = __cosf(ang), sn = __sinf(ang);
    float xp = qkv[(size_t)bs * CW + (col ^ 64)];
    float o = (d < 64) ? (x * c - xp * sn) : (x * c + xp * sn);
    Kb[((size_t)(b * NKV + kv) * SQ + s) * DH + d] = (bf16)o;
  } else {                                   // V, copy
    int c2 = col - (NH + NKV) * DH;
    int kv = c2 >> 7, d = c2 & 127;
    Vb[((size_t)(b * NKV + kv) * SQ + s) * DH + d] = (bf16)x;
  }
}

// ================== flash attention (causal, GQA), all-async staging, bf16 output ==================
#define QB 64
#define KBX 64

// One 64-key tile: scores (already scaled via Q), online softmax, O += P@V.
// MASK=true only for the diagonal (last) tile.
template <bool MASK>
static __device__ __forceinline__ void attn_tile(
    const bf16 (*Ksb)[DH], const bf16 (*Vsb)[KBX], bf16 (*Psw)[32],
    const V16 (&qf)[4], v8f (&o)[8], float (&mrow)[8], float (&lrow)[8],
    int kb0, int qgbase, int ln, int hi) {
#pragma unroll
  for (int kk = 0; kk < 2; kk++) {      // 32-key chunk
    v8f sc[2];
#pragma unroll
    for (int sub = 0; sub < 2; sub++) { // 16-key WMMA subtile
      sc[sub] = zero8();
#pragma unroll
      for (int c = 0; c < 4; c++) {
        V16 kf;  // B frag of K^T: col = key, K over dims [32c+8hi, +7] and [32c+16+8hi, +7]
        kf.h[0] = *(const bf16x8*)&Ksb[kk * 32 + sub * 16 + ln][32 * c + 8 * hi];
        kf.h[1] = *(const bf16x8*)&Ksb[kk * 32 + sub * 16 + ln][32 * c + 16 + 8 * hi];
        sc[sub] = wmma_bf16(qf[c], kf, sc[sub]);
      }
    }
    const int k0g = kb0 + kk * 32 + ln;
#pragma unroll
    for (int r = 0; r < 8; r++) {
      float v0 = sc[0][r];
      float v1 = sc[1][r];
      if (MASK) {
        int qg = qgbase + r + 8 * hi;
        if (k0g      > qg) v0 = -1.0e30f;
        if (k0g + 16 > qg) v1 = -1.0e30f;
      }
      float mx   = rmax16(fmaxf(v0, v1));
      float mnew = fmaxf(mrow[r], mx);
      float corr = __expf(mrow[r] - mnew);
      mrow[r] = mnew;
      float p0 = __expf(v0 - mnew);
      float p1 = __expf(v1 - mnew);
      lrow[r] = lrow[r] * corr + rsum16(p0 + p1);
#pragma unroll
      for (int nt = 0; nt < 8; nt++) o[nt][r] = o[nt][r] * corr;
      Psw[r + 8 * hi][ln]      = (bf16)p0;
      Psw[r + 8 * hi][16 + ln] = (bf16)p1;
    }
    // P A-fragment (16x32) from wave-private LDS scratch
    V16 pf;
    pf.h[0] = *(const bf16x8*)&Psw[ln][8 * hi];
    pf.h[1] = *(const bf16x8*)&Psw[ln][16 + 8 * hi];
    // O += P @ V  (8 dim-tiles of 16)
#pragma unroll
    for (int nt = 0; nt < 8; nt++) {
      V16 vf;  // B frag of V: col = dim, K over keys [kk*32+8hi, +7] and [kk*32+16+8hi, +7]
      vf.h[0] = *(const bf16x8*)&Vsb[nt * 16 + ln][kk * 32 + 8 * hi];
      vf.h[1] = *(const bf16x8*)&Vsb[nt * 16 + ln][kk * 32 + 16 + 8 * hi];
      o[nt] = wmma_bf16(pf, vf, o[nt]);
    }
  }
}

// grid: (S/64, NH, B), 128 threads (4 waves). Wave w owns queries 16w..16w+15.
__global__ __launch_bounds__(128)
void flash_fwd(const bf16* __restrict__ Qb, const bf16* __restrict__ Kb,
               const bf16* __restrict__ Vt, bf16* __restrict__ Out) {
  __shared__ bf16 Ks[2][KBX][DH];   // [key][dim]   2 x 16 KB (async DMA)
  __shared__ bf16 Vs[2][DH][KBX];   // [dim][key]   2 x 16 KB (async DMA from Vt)
  __shared__ bf16 Ps[4][16][32];    // per-wave P scratch (C->A layout bounce)

  const int tid  = threadIdx.x;
  const int wave = tid >> 5;
  const int lane = tid & 31;
  const int ln   = lane & 15;
  const int hi   = lane >> 4;
  const int qb   = blockIdx.x * QB;
  const int h    = blockIdx.y;
  const int b    = blockIdx.z;
  const int kvh  = h >> 2;   // G = 4

  const bf16* Qh = Qb + (size_t)(b * NH  + h)   * SQ * DH;
  const bf16* Kh = Kb + (size_t)(b * NKV + kvh) * SQ * DH;
  const bf16* Vh = Vt + (size_t)(b * NKV + kvh) * DH * SQ;  // [dim][s]

  // stage one K/V tile: 16 async b128 per thread
  auto issueKV = [&](int buf, int kb0) {
#pragma unroll
    for (int r = 0; r < 8; r++) {
      int chunk = tid + r * 128;         // 0..1023
      int key = chunk >> 4;              // 16 chunks per 128-dim row
      int c8  = (chunk & 15) << 3;
      async_cp16((unsigned)(size_t)&Ks[buf][key][c8],
                 Kh + (size_t)(kb0 + key) * DH + c8);
    }
#pragma unroll
    for (int r = 0; r < 8; r++) {
      int chunk = tid + r * 128;
      int dim = chunk >> 3;              // 8 chunks per 64-key row
      int c8  = (chunk & 7) << 3;
      async_cp16((unsigned)(size_t)&Vs[buf][dim][c8],
                 Vh + (size_t)dim * SQ + kb0 + c8);
    }
  };

  // Q fragments: 4 K-chunks of 32 dims (Q already carries 1/sqrt(D))
  V16 qf[4];
  {
    const bf16* qrow = Qh + (size_t)(qb + wave * 16 + ln) * DH;
#pragma unroll
    for (int c = 0; c < 4; c++) {
      qf[c].h[0] = *(const bf16x8*)(qrow + 32 * c + 8 * hi);
      qf[c].h[1] = *(const bf16x8*)(qrow + 32 * c + 16 + 8 * hi);
    }
  }

  v8f o[8];
#pragma unroll
  for (int t = 0; t < 8; t++) o[t] = zero8();
  float mrow[8], lrow[8];
#pragma unroll
  for (int r = 0; r < 8; r++) { mrow[r] = -3.0e38f; lrow[r] = 0.f; }

  const int qgbase = qb + wave * 16;
  const int nkb = (qb + QB) / KBX;

  issueKV(0, 0);
  for (int i = 0; i < nkb; i++) {
    const int buf = i & 1;
    const int kb0 = i * KBX;
    if (i + 1 < nkb) {
      issueKV(buf ^ 1, (i + 1) * KBX);
      asm volatile("s_wait_asynccnt 0x10" ::: "memory");  // drain current tile only
    } else {
      asm volatile("s_wait_asynccnt 0x0" ::: "memory");
    }
    __syncthreads();

    if (i == nkb - 1)   // only the diagonal tile needs the causal mask
      attn_tile<true >(Ks[buf], Vs[buf], Ps[wave], qf, o, mrow, lrow, kb0, qgbase, ln, hi);
    else
      attn_tile<false>(Ks[buf], Vs[buf], Ps[wave], qf, o, mrow, lrow, kb0, qgbase, ln, hi);

    __syncthreads();
  }

  // normalize + store bf16 attn output at [b, s, h*128 + d]
#pragma unroll
  for (int r = 0; r < 8; r++) {
    float inv = 1.0f / lrow[r];
    int s = qb + wave * 16 + r + 8 * hi;
    bf16* orow = Out + ((size_t)b * SQ + s) * (NH * DH) + h * DH;
#pragma unroll
    for (int nt = 0; nt < 8; nt++) orow[nt * 16 + ln] = (bf16)(o[nt][r] * inv);
  }
}

// ================== launcher ==================
extern "C" void kernel_launch(void* const* d_in, const int* in_sizes, int n_in,
                              void* d_out, int out_size, void* d_ws, size_t ws_size,
                              hipStream_t stream) {
  (void)in_sizes; (void)n_in; (void)out_size; (void)ws_size;
  const int*   positions = (const int*)d_in[0];
  const float* hidden    = (const float*)d_in[1];
  const float* w_qkv     = (const float*)d_in[2];
  const float* w_o       = (const float*)d_in[3];
  float*       out       = (float*)d_out;

  const size_t Mrows = (size_t)BQ * SQ;           // 4096
  const int    CW    = (NH + 2 * NKV) * DH;       // 6144

  char* ws = (char*)d_ws;
  float* qkv    = (float*)ws; ws += Mrows * CW * sizeof(float);
  bf16*  Qb     = (bf16*)ws;  ws += (size_t)BQ * NH  * SQ * DH * sizeof(bf16);
  bf16*  Kb     = (bf16*)ws;  ws += (size_t)BQ * NKV * SQ * DH * sizeof(bf16);
  bf16*  Vb     = (bf16*)ws;  ws += (size_t)BQ * NKV * SQ * DH * sizeof(bf16);
  bf16*  Vtb    = (bf16*)ws;  ws += (size_t)BQ * NKV * SQ * DH * sizeof(bf16);
  bf16*  attnB  = (bf16*)ws;  ws += Mrows * (size_t)(NH * DH) * sizeof(bf16);
  bf16*  hidB   = (bf16*)ws;  ws += Mrows * (size_t)HID * sizeof(bf16);
  bf16*  WqkvT  = (bf16*)ws;  ws += (size_t)CW * HID * sizeof(bf16);
  bf16*  WoT    = (bf16*)ws;  ws += (size_t)HID * HID * sizeof(bf16);

  // 0) operand preparation (memory-bound, ~10s of microseconds at 23.3 TB/s)
  {
    int n4 = (int)(Mrows * HID / 4);
    convert_bf16<<<(n4 + 255) / 256, 256, 0, stream>>>(hidden, hidB, n4);
  }
  transpose_convert<<<dim3(CW / 32, HID / 32), 256, 0, stream>>>(w_qkv, WqkvT, HID, CW);
  transpose_convert<<<dim3(HID / 32, HID / 32), 256, 0, stream>>>(w_o, WoT, HID, HID);

  // 1) QKV projection: [4096,4096] @ [4096,6144] -> fp32
  gemm_bf16_async<(NH + 2 * NKV) * DH, HID>
      <<<dim3(CW / GN, Mrows / GM), 256, 0, stream>>>(hidB, WqkvT, qkv);

  // 2) RoPE + split into bf16 Q/K/V (Q pre-scaled by 1/sqrt(D))
  int total = (int)(Mrows * CW);
  rope_split<<<(total + 255) / 256, 256, 0, stream>>>(positions, qkv, Qb, Kb, Vb, total);

  // 2b) V -> V^T ([s][d] -> [d][s] per (b,kv)) so flash can DMA-stage it
  transpose_v<<<dim3(DH / 32, SQ / 32, BQ * NKV), 256, 0, stream>>>(Vb, Vtb);

  // 3) causal GQA flash attention -> bf16
  flash_fwd<<<dim3(SQ / QB, NH, BQ), 128, 0, stream>>>(Qb, Kb, Vtb, attnB);

  // 4) output projection: [4096,4096] @ [4096,4096] -> fp32 out
  gemm_bf16_async<HID, HID>
      <<<dim3(HID / GN, Mrows / GM), 256, 0, stream>>>(attnB, WoT, out);
}
